// TalkingHeadAttn_78984448573776
// MI455X (gfx1250) — matrix operand order repaired and checked by reading, not compiled, not run on hardware
//
#include <hip/hip_runtime.h>
#include <hip/hip_bf16.h>

// ---------------------------------------------------------------------------
// Talking-heads attention for MI455X (gfx1250), wave32 + WMMA bf16.
// B=16, N=576, C=768, H=12, HD=64, SCALE=1/8
// ---------------------------------------------------------------------------

#define B_   16
#define N_   576
#define C_   768
#define H_   12
#define HD_  64
#define C3_  2304
#define MTOK 9216          // B*N
#define SCALE_F 0.125f

using bf16 = __bf16;
typedef __attribute__((ext_vector_type(8)))  bf16  v8bf;
typedef __attribute__((ext_vector_type(16))) bf16  v16bf;
typedef __attribute__((ext_vector_type(8)))  float v8f;

// ---------------- WMMA helpers -------------------------------------------

__device__ inline v8f zero8() {
    v8f z;
#pragma unroll
    for (int i = 0; i < 8; ++i) z[i] = 0.0f;
    return z;
}

// A-matrix 16x32 bf16 fragment (row-major, K-contiguous, row stride ldk elems)
// lane L (<16): row=L,   K chunks [0..7]  and [16..23]
// lane L (>=16):row=L-16,K chunks [8..15] and [24..31]
__device__ inline v16bf ld_frag_a(const bf16* __restrict__ base, int ldk) {
    const int lane = threadIdx.x & 31;
    const bf16* r = base + (lane & 15) * ldk + ((lane >> 4) << 3);
    v8bf lo = *(const v8bf*)(r);
    v8bf hi = *(const v8bf*)(r + 16);
    v16bf out;
#pragma unroll
    for (int i = 0; i < 8; ++i) { out[i] = lo[i]; out[i + 8] = hi[i]; }
    return out;
}

// B-matrix 32x16 bf16 fragment; operand stored as N rows x K cols (K-contig,
// row stride ldk). lane L: col=L&15, K chunks (L>>4)*8 .. +7 and 16+(L>>4)*8 .. +7
__device__ inline v16bf ld_frag_b(const bf16* __restrict__ base, int ldk) {
    const int lane = threadIdx.x & 31;
    const bf16* r = base + (lane & 15) * ldk + ((lane >> 4) << 3);
    v8bf lo = *(const v8bf*)(r);
    v8bf hi = *(const v8bf*)(r + 16);
    v16bf out;
#pragma unroll
    for (int i = 0; i < 8; ++i) { out[i] = lo[i]; out[i + 8] = hi[i]; }
    return out;
}

__device__ inline v8f wmma_bf16(v16bf a, v16bf b, v8f c) {
    return __builtin_amdgcn_wmma_f32_16x16x32_bf16(
        /*neg_a=*/false, a, /*neg_b=*/false, b,
        /*c_mod=*/(short)0, c, /*reuse_a=*/false, /*reuse_b=*/false);
}

// ---------------- Kernel 0: fp32 -> bf16 convert --------------------------

__global__ __launch_bounds__(256) void f2bf_kernel(const float* __restrict__ s,
                                                   bf16* __restrict__ d, int n) {
    int i = blockIdx.x * blockDim.x + threadIdx.x;
    int stride = gridDim.x * blockDim.x;
    for (; i < n; i += stride) d[i] = (bf16)s[i];
}

// ---------------- Kernel 1: QKV projection --------------------------------
// Y = Xb(9216x768) @ Wqkv(2304x768)^T ; scatter into Q(B,H,N,HD)*SCALE,
// K(B,H,N,HD), Vt(B,H,HD,N) all bf16.

__global__ __launch_bounds__(256) void qkv_gemm(const bf16* __restrict__ Xb,
                                                const bf16* __restrict__ Wb,
                                                bf16* __restrict__ Qb,
                                                bf16* __restrict__ Kb,
                                                bf16* __restrict__ Vt) {
    const int wave = blockIdx.x * (blockDim.x >> 5) + (threadIdx.x >> 5);
    const int NT = C3_ / 64;                 // 36
    const int mt = wave / NT, nt = wave % NT;
    if (mt >= MTOK / 64) return;
    const int m0 = mt * 64, n0 = nt * 64;

    v8f acc[4][4];
#pragma unroll
    for (int i = 0; i < 4; ++i)
#pragma unroll
        for (int j = 0; j < 4; ++j) acc[i][j] = zero8();

    for (int k = 0; k < C_; k += 32) {
        v16bf a[4], b[4];
#pragma unroll
        for (int i = 0; i < 4; ++i)
            a[i] = ld_frag_a(Xb + (size_t)(m0 + 16 * i) * C_ + k, C_);
#pragma unroll
        for (int j = 0; j < 4; ++j)
            b[j] = ld_frag_b(Wb + (size_t)(n0 + 16 * j) * C_ + k, C_);
#pragma unroll
        for (int i = 0; i < 4; ++i)
#pragma unroll
            for (int j = 0; j < 4; ++j)
                acc[i][j] = wmma_bf16(a[i], b[j], acc[i][j]);
    }

    const int lane = threadIdx.x & 31;
    const int col = lane & 15, rbase = (lane >> 4) * 8;
#pragma unroll
    for (int i = 0; i < 4; ++i) {
#pragma unroll
        for (int j = 0; j < 4; ++j) {
#pragma unroll
            for (int r = 0; r < 8; ++r) {
                const int m = m0 + 16 * i + rbase + r;
                const int o = n0 + 16 * j + col;
                const int bb = m / N_, tok = m % N_;
                const int s = o / C_, oo = o % C_;
                const int h = oo >> 6, d = oo & 63;
                const float v = acc[i][j][r];
                if (s == 0)
                    Qb[(((size_t)bb * H_ + h) * N_ + tok) * HD_ + d] = (bf16)(v * SCALE_F);
                else if (s == 1)
                    Kb[(((size_t)bb * H_ + h) * N_ + tok) * HD_ + d] = (bf16)v;
                else
                    Vt[(((size_t)bb * H_ + h) * HD_ + d) * N_ + tok] = (bf16)v;
            }
        }
    }
}

// ---------------- Kernel 2: raw scores S = Q @ K^T ------------------------
// per (b,h): (576x64)@(64x576) -> S(B,H,N,N) bf16

__global__ __launch_bounds__(256) void scores_gemm(const bf16* __restrict__ Qb,
                                                   const bf16* __restrict__ Kb,
                                                   bf16* __restrict__ S) {
    const int wave = blockIdx.x * (blockDim.x >> 5) + (threadIdx.x >> 5);
    const int TPB = 81;                      // (576/64)^2 wave-tiles per (b,h)
    const int bh = wave / TPB, t = wave % TPB;
    if (bh >= B_ * H_) return;
    const int m0 = (t / 9) * 64, n0 = (t % 9) * 64;

    const bf16* q = Qb + (size_t)bh * N_ * HD_;
    const bf16* kk = Kb + (size_t)bh * N_ * HD_;

    v8f acc[4][4];
#pragma unroll
    for (int i = 0; i < 4; ++i)
#pragma unroll
        for (int j = 0; j < 4; ++j) acc[i][j] = zero8();

#pragma unroll
    for (int k = 0; k < HD_; k += 32) {
        v16bf a[4], b[4];
#pragma unroll
        for (int i = 0; i < 4; ++i)
            a[i] = ld_frag_a(q + (size_t)(m0 + 16 * i) * HD_ + k, HD_);
#pragma unroll
        for (int j = 0; j < 4; ++j)
            b[j] = ld_frag_b(kk + (size_t)(n0 + 16 * j) * HD_ + k, HD_);
#pragma unroll
        for (int i = 0; i < 4; ++i)
#pragma unroll
            for (int j = 0; j < 4; ++j)
                acc[i][j] = wmma_bf16(a[i], b[j], acc[i][j]);
    }

    bf16* so = S + (size_t)bh * N_ * N_;
    const int lane = threadIdx.x & 31;
    const int col = lane & 15, rbase = (lane >> 4) * 8;
#pragma unroll
    for (int i = 0; i < 4; ++i)
#pragma unroll
        for (int j = 0; j < 4; ++j)
#pragma unroll
            for (int r = 0; r < 8; ++r)
                so[(size_t)(m0 + 16 * i + rbase + r) * N_ + n0 + 16 * j + col] =
                    (bf16)acc[i][j][r];
}

// ---------------- Kernel 3: fused head-mix -> softmax -> head-mix ----------
// one block per (b,m) row; 576 threads (one per key n)

__device__ inline float blk_reduce_max(float v, float* red, int n) {
    red[n] = v; __syncthreads();
    if (n < 64) red[n] = fmaxf(red[n], red[n + 512]);
    __syncthreads();
    for (int s = 256; s >= 1; s >>= 1) {
        if (n < s) red[n] = fmaxf(red[n], red[n + s]);
        __syncthreads();
    }
    float r = red[0]; __syncthreads();
    return r;
}

__device__ inline float blk_reduce_sum(float v, float* red, int n) {
    red[n] = v; __syncthreads();
    if (n < 64) red[n] += red[n + 512];
    __syncthreads();
    for (int s = 256; s >= 1; s >>= 1) {
        if (n < s) red[n] += red[n + s];
        __syncthreads();
    }
    float r = red[0]; __syncthreads();
    return r;
}

__global__ __launch_bounds__(576) void mix_softmax_mix(const bf16* __restrict__ S,
                                                       const float* __restrict__ Wl,
                                                       const float* __restrict__ bl,
                                                       const float* __restrict__ Ww,
                                                       const float* __restrict__ bw,
                                                       bf16* __restrict__ P) {
    __shared__ float Sl[H_][N_];   // raw scores, all heads, this row
    __shared__ float Tl[H_][N_];   // mixed scores -> probabilities
    __shared__ float Wls[H_ * H_], Wws[H_ * H_], bls[H_], bws[H_];
    __shared__ float red[640];

    const int bm = blockIdx.x;           // b*N + m
    const int n = threadIdx.x;
    const int b = bm / N_, m = bm % N_;

    const bf16* srow = S + ((size_t)b * H_) * N_ * N_ + (size_t)m * N_;
#pragma unroll
    for (int h = 0; h < H_; ++h) Sl[h][n] = (float)srow[(size_t)h * N_ * N_ + n];
    if (n < H_ * H_) { Wls[n] = Wl[n]; Wws[n] = Ww[n]; }
    if (n < H_)      { bls[n] = bl[n]; bws[n] = bw[n]; }
    __syncthreads();

    // pre-softmax talking-heads mix: T[g] = bl[g] + sum_h Wl[g,h]*S[h]
#pragma unroll
    for (int g = 0; g < H_; ++g) {
        float a = bls[g];
#pragma unroll
        for (int h = 0; h < H_; ++h) a += Wls[g * H_ + h] * Sl[h][n];
        Tl[g][n] = a;
    }
    __syncthreads();

    // softmax each head row
    for (int g = 0; g < H_; ++g) {
        float v = Tl[g][n];
        float mx = blk_reduce_max(v, red, n);
        float e = __expf(v - mx);
        float sm = blk_reduce_sum(e, red, n);
        Tl[g][n] = e / sm;
        __syncthreads();
    }

    // post-softmax mix: P[g] = bw[g] + sum_h Ww[g,h]*prob[h]
    bf16* prow = P + ((size_t)b * H_) * N_ * N_ + (size_t)m * N_;
#pragma unroll
    for (int g = 0; g < H_; ++g) {
        float a = bws[g];
#pragma unroll
        for (int h = 0; h < H_; ++h) a += Wws[g * H_ + h] * Tl[h][n];
        prow[(size_t)g * N_ * N_ + n] = (bf16)a;
    }
}

// ---------------- Kernel 4: O = P @ V  ------------------------------------
// per (b,h): (576x576)@(576x64) with Vt stored (HD,N); out -> AO(B,N,C) bf16

__global__ __launch_bounds__(256) void pv_gemm(const bf16* __restrict__ P,
                                               const bf16* __restrict__ Vt,
                                               bf16* __restrict__ AO) {
    const int wave = blockIdx.x * (blockDim.x >> 5) + (threadIdx.x >> 5);
    const int bh = wave / 9, mt = wave % 9;
    if (bh >= B_ * H_) return;
    const int m0 = mt * 64;
    const int b = bh / H_, h = bh % H_;

    const bf16* p = P + (size_t)bh * N_ * N_;
    const bf16* vt = Vt + (size_t)bh * HD_ * N_;

    v8f acc[4][4];
#pragma unroll
    for (int i = 0; i < 4; ++i)
#pragma unroll
        for (int j = 0; j < 4; ++j) acc[i][j] = zero8();

    for (int k = 0; k < N_; k += 32) {
        v16bf a[4], bfr[4];
#pragma unroll
        for (int i = 0; i < 4; ++i)
            a[i] = ld_frag_a(p + (size_t)(m0 + 16 * i) * N_ + k, N_);
#pragma unroll
        for (int j = 0; j < 4; ++j)
            bfr[j] = ld_frag_b(vt + (size_t)(16 * j) * N_ + k, N_);
#pragma unroll
        for (int i = 0; i < 4; ++i)
#pragma unroll
            for (int j = 0; j < 4; ++j)
                acc[i][j] = wmma_bf16(a[i], bfr[j], acc[i][j]);
    }

    const int lane = threadIdx.x & 31;
    const int col = lane & 15, rbase = (lane >> 4) * 8;
#pragma unroll
    for (int i = 0; i < 4; ++i)
#pragma unroll
        for (int j = 0; j < 4; ++j)
#pragma unroll
            for (int r = 0; r < 8; ++r) {
                const int tok = m0 + 16 * i + rbase + r;
                const int d = 16 * j + col;
                AO[((size_t)b * N_ + tok) * C_ + h * HD_ + d] = (bf16)acc[i][j][r];
            }
}

// ---------------- Kernel 5: output projection ------------------------------
// out = AO(9216x768) @ Wproj(768x768)^T + bproj ; fp32 output

__global__ __launch_bounds__(256) void proj_gemm(const bf16* __restrict__ AO,
                                                 const bf16* __restrict__ Wb,
                                                 const float* __restrict__ bias,
                                                 float* __restrict__ out) {
    const int wave = blockIdx.x * (blockDim.x >> 5) + (threadIdx.x >> 5);
    const int NT = C_ / 64;                  // 12
    const int mt = wave / NT, nt = wave % NT;
    if (mt >= MTOK / 64) return;
    const int m0 = mt * 64, n0 = nt * 64;

    v8f acc[4][4];
#pragma unroll
    for (int i = 0; i < 4; ++i)
#pragma unroll
        for (int j = 0; j < 4; ++j) acc[i][j] = zero8();

    for (int k = 0; k < C_; k += 32) {
        v16bf a[4], b[4];
#pragma unroll
        for (int i = 0; i < 4; ++i)
            a[i] = ld_frag_a(AO + (size_t)(m0 + 16 * i) * C_ + k, C_);
#pragma unroll
        for (int j = 0; j < 4; ++j)
            b[j] = ld_frag_b(Wb + (size_t)(n0 + 16 * j) * C_ + k, C_);
#pragma unroll
        for (int i = 0; i < 4; ++i)
#pragma unroll
            for (int j = 0; j < 4; ++j)
                acc[i][j] = wmma_bf16(a[i], b[j], acc[i][j]);
    }

    const int lane = threadIdx.x & 31;
    const int col = lane & 15, rbase = (lane >> 4) * 8;
#pragma unroll
    for (int i = 0; i < 4; ++i)
#pragma unroll
        for (int j = 0; j < 4; ++j) {
            const int o = n0 + 16 * j + col;
            const float bo = bias[o];
#pragma unroll
            for (int r = 0; r < 8; ++r) {
                const int m = m0 + 16 * i + rbase + r;
                out[(size_t)m * C_ + o] = acc[i][j][r] + bo;
            }
        }
}

// ---------------- host-side launch -----------------------------------------

extern "C" void kernel_launch(void* const* d_in, const int* in_sizes, int n_in,
                              void* d_out, int out_size, void* d_ws, size_t ws_size,
                              hipStream_t stream) {
    const float* x     = (const float*)d_in[0];
    const float* Wqkv  = (const float*)d_in[1];
    const float* Wl    = (const float*)d_in[2];
    const float* bl    = (const float*)d_in[3];
    const float* Ww    = (const float*)d_in[4];
    const float* bw    = (const float*)d_in[5];
    const float* Wproj = (const float*)d_in[6];
    const float* bproj = (const float*)d_in[7];
    float* out = (float*)d_out;

    // workspace carve-up (bf16 buffers, all 256B-aligned sizes)
    char* ws = (char*)d_ws;
    size_t off = 0;
    bf16* Xb   = (bf16*)(ws + off); off += (size_t)MTOK * C_ * 2;          // 14.2 MB
    bf16* Wqb  = (bf16*)(ws + off); off += (size_t)C3_ * C_ * 2;           //  3.5 MB
    bf16* Wpb  = (bf16*)(ws + off); off += (size_t)C_ * C_ * 2;            //  1.2 MB
    bf16* Qb   = (bf16*)(ws + off); off += (size_t)B_ * H_ * N_ * HD_ * 2; // 14.2 MB
    bf16* Kb   = (bf16*)(ws + off); off += (size_t)B_ * H_ * N_ * HD_ * 2; // 14.2 MB
    bf16* Vt   = (bf16*)(ws + off); off += (size_t)B_ * H_ * N_ * HD_ * 2; // 14.2 MB
    bf16* S    = (bf16*)(ws + off); off += (size_t)B_ * H_ * N_ * N_ * 2;  // 127 MB
    bf16* P    = (bf16*)(ws + off); off += (size_t)B_ * H_ * N_ * N_ * 2;  // 127 MB
    bf16* AO   = (bf16*)(ws + off); off += (size_t)MTOK * C_ * 2;          // 14.2 MB

    // 0: fp32 -> bf16 conversions
    f2bf_kernel<<<2048, 256, 0, stream>>>(x, Xb, MTOK * C_);
    f2bf_kernel<<<2048, 256, 0, stream>>>(Wqkv, Wqb, C3_ * C_);
    f2bf_kernel<<<1024, 256, 0, stream>>>(Wproj, Wpb, C_ * C_);

    // 1: QKV projection (144*36 = 5184 wave-tiles, 8 waves/block)
    qkv_gemm<<<648, 256, 0, stream>>>(Xb, Wqb, Qb, Kb, Vt);

    // 2: raw attention scores (192*81 = 15552 wave-tiles)
    scores_gemm<<<1944, 256, 0, stream>>>(Qb, Kb, S);

    // 3: fused talking-heads mix + softmax + mix (one block per (b,m) row)
    mix_softmax_mix<<<B_ * N_, 576, 0, stream>>>(S, Wl, bl, Ww, bw, P);

    // 4: probs @ V (192*9 = 1728 wave-tiles)
    pv_gemm<<<216, 256, 0, stream>>>(P, Vt, AO);

    // 5: output projection + bias (144*12 = 1728 wave-tiles)
    proj_gemm<<<216, 256, 0, stream>>>(AO, Wpb, bproj, out);

    (void)in_sizes; (void)n_in; (void)out_size; (void)ws_size;
}
